// CombineGraph_21586505630397
// MI455X (gfx1250) — compile-verified, matrix operand order checked by reading, not run on hardware
//
#include <hip/hip_runtime.h>

// ---------- types ----------
typedef float  v8f   __attribute__((ext_vector_type(8)));
typedef __bf16 v16bf __attribute__((ext_vector_type(16)));
typedef __bf16 v8bf  __attribute__((ext_vector_type(8)));

#define B_SZ   128
#define L_SZ   40
#define D_SZ   128
#define S_NB   12
#define HOP_N  2
#define NEGBIG (-9e15f)

// ---------- CDNA5 async global->LDS copy (ASYNCcnt-tracked, no VGPR data) ----
typedef __attribute__((address_space(3))) void as3_void;

__device__ __forceinline__ void async_copy_b128(const void* g, void* l) {
    unsigned lds_off = (unsigned)(unsigned long long)(as3_void*)l;
    unsigned long long ga = (unsigned long long)g;
    asm volatile("global_load_async_to_lds_b128 %0, %1, off"
                 :: "v"(lds_off), "v"(ga)
                 : "memory");
}
__device__ __forceinline__ void wait_async0() {
    asm volatile("s_wait_asynccnt 0x0" ::: "memory");
}
__device__ __forceinline__ void wait_ds0() {
    asm volatile("s_wait_dscnt 0x0" ::: "memory");
}

// ===========================================================================
// Pre-swizzle weight matrices W [HOP][K][128] (f32, row-major) into bf16 WMMA
// B-fragments: layout [hop][kc][t][lane][16]; lane's 16 elems are two
// contiguous 8-element K runs (16-bit operand VGPR layout for 16x16x32).
// ===========================================================================
__global__ void frag_pack_kernel(const float* __restrict__ W,
                                 __bf16* __restrict__ outp,
                                 int kchunks, int hop_stride, int total) {
    int tid = blockIdx.x * blockDim.x + threadIdx.x;
    if (tid >= total) return;
    int lane = tid & 31;
    int t    = (tid >> 5) & 7;
    int kc   = (tid >> 8) % kchunks;
    int hop  = tid / (256 * kchunks);
    const float* w = W + (size_t)hop * hop_stride;
    int n       = t * 16 + (lane & 15);
    int base_lo = kc * 32 + ((lane & 16) ? 8 : 0);
    __bf16* o = outp + (size_t)tid * 16;
#pragma unroll
    for (int e = 0; e < 8; ++e) {
        o[e]     = (__bf16)w[(base_lo + e) * D_SZ + n];
        o[e + 8] = (__bf16)w[(base_lo + 16 + e) * D_SZ + n];
    }
}

// ===========================================================================
// sum_item_emb: [B,D] = sum_l emb[item[b,l]]*mask / mask.sum ; also masksum[b]
// ===========================================================================
__global__ void sum_item_kernel(const float* __restrict__ emb,
                                const int* __restrict__ item,
                                const int* __restrict__ mask,
                                float* __restrict__ sumit,
                                float* __restrict__ msum) {
    int b = blockIdx.x, d = threadIdx.x;
    float acc = 0.f, ms = 0.f;
    for (int l = 0; l < L_SZ; ++l) {
        float mf = (float)mask[b * L_SZ + l];
        acc += emb[(size_t)item[b * L_SZ + l] * D_SZ + d] * mf;
        ms  += mf;
    }
    sumit[b * D_SZ + d] = acc / ms;
    if (d == 0) msum[b] = ms;
}

// ===========================================================================
// expand: out_idx[i] = adj_all[src[i/S]*S + i%S]; out_w likewise from num_tab
// ===========================================================================
__global__ void expand_kernel(const int* __restrict__ src,
                              const int* __restrict__ adj_all,
                              const float* __restrict__ num_tab,
                              int* __restrict__ out_idx,
                              float* __restrict__ out_w, int count) {
    int tid = blockIdx.x * blockDim.x + threadIdx.x;
    if (tid >= count) return;
    int s = src[tid / S_NB];
    int c = tid % S_NB;
    out_idx[tid] = adj_all[(size_t)s * S_NB + c];
    out_w[tid]   = num_tab[(size_t)s * S_NB + c];
}

// ===========================================================================
// Local aggregator (small, VALU): h_local written straight into d_out.
// ===========================================================================
__global__ __launch_bounds__(256) void local_agg_kernel(
    const float* __restrict__ emb, const int* __restrict__ inputs,
    const int* __restrict__ adj, const float* __restrict__ a_loc,
    float* __restrict__ outp) {
    __shared__ float sh_h[L_SZ][D_SZ];
    __shared__ float sh_a[D_SZ][4];
    __shared__ float sh_att[L_SZ][L_SZ];
    const int b = blockIdx.x, t = threadIdx.x;

    for (int q = t; q < L_SZ * D_SZ; q += 256) {
        int i = q >> 7, d = q & 127;
        sh_h[i][d] = emb[(size_t)inputs[b * L_SZ + i] * D_SZ + d];
    }
    ((float*)sh_a)[t]       = a_loc[t];
    ((float*)sh_a)[t + 256] = a_loc[t + 256];
    __syncthreads();

    for (int p = t; p < L_SZ * L_SZ; p += 256) {
        int i = p / L_SZ, j = p % L_SZ;
        float e0 = 0, e1 = 0, e2 = 0, e3 = 0;
        for (int d = 0; d < D_SZ; ++d) {
            float hij = sh_h[i][d] * sh_h[j][d];
            e0 += hij * sh_a[d][0];
            e1 += hij * sh_a[d][1];
            e2 += hij * sh_a[d][2];
            e3 += hij * sh_a[d][3];
        }
        int code = adj[(size_t)b * (L_SZ * L_SZ) + p];
        float es[4] = {e0, e1, e2, e3};
        float lg = NEGBIG;
        if (code >= 1 && code <= 4) {
            float v = es[code - 1];
            lg = (v > 0.f) ? v : 0.2f * v;
        }
        sh_att[i][j] = lg;
    }
    __syncthreads();

    const int wv = t >> 5, ln = t & 31;
    for (int i = wv; i < L_SZ; i += 8) {
        float v1 = sh_att[i][ln];
        float v2 = (ln < 8) ? sh_att[i][32 + ln] : -3.4e38f;
        float mx = fmaxf(v1, v2);
        for (int o = 1; o < 32; o <<= 1) mx = fmaxf(mx, __shfl_xor(mx, o, 32));
        float x1 = __expf(v1 - mx);
        float x2 = (ln < 8) ? __expf(v2 - mx) : 0.f;
        float sm = x1 + x2;
        for (int o = 1; o < 32; o <<= 1) sm += __shfl_xor(sm, o, 32);
        float inv = 1.f / sm;
        sh_att[i][ln] = x1 * inv;
        if (ln < 8) sh_att[i][32 + ln] = x2 * inv;
    }
    __syncthreads();

    for (int q = t; q < L_SZ * D_SZ; q += 256) {
        int i = q >> 7, d = q & 127;
        float acc = 0.f;
#pragma unroll
        for (int j = 0; j < L_SZ; ++j) acc += sh_att[i][j] * sh_h[j][d];
        outp[(size_t)b * (L_SZ * D_SZ) + q] = acc;
    }
}

// ===========================================================================
// Global aggregator. 128 threads = 4 waves; workgroup owns 16 (b,i) rows.
// Phase 1 (per wave, 4 rows sequentially): neighbor rows staged with
//   global_load_async_to_lds_b128 (ASYNCcnt), then attention via
//   v_wmma_f32_16x16x32_bf16  [16(M=12 pad) x 128] @ w1[128 x 128] (+bias row)
// Phase 2 (whole WG): [16 x 256] @ w3[256 x 128] via WMMA, ReLU, store.
// Embedding rows gathered directly (table is L2-resident, 25.6MB << 192MB).
// Tile loops kept at unroll 1 to stay well under 256 VGPRs (occupancy).
// ===========================================================================
__global__ __launch_bounds__(128) void global_agg_kernel(
    const float* __restrict__ self_base, const int* __restrict__ self_idx,
    const float* __restrict__ neigh_base, const int* __restrict__ neigh_idx,
    const float* __restrict__ wgt,      // [rows*S]
    const float* __restrict__ extra,    // [B,128] sum_item_emb
    const __bf16* __restrict__ w1frag,  // [4][8][32][16]
    const float* __restrict__ w1last,   // w1 row K=128 (bias), [128]
    const float* __restrict__ w2v,      // [128]
    const __bf16* __restrict__ w3frag,  // [8][8][32][16]
    float* __restrict__ outp,           // [rows*128]
    int rows_per_b) {
    __shared__ float sh_neigh[4][16][D_SZ];
    __shared__ __align__(16) __bf16 sh_concat[16][2 * D_SZ];
    __shared__ float sh_extra[4][D_SZ];
    __shared__ float sh_wrow[4][16];
    __shared__ float sh_w2[D_SZ];
    __shared__ float sh_w1l[D_SZ];

    const int tid  = threadIdx.x;
    const int wave = tid >> 5;
    const int lane = tid & 31;
    const int mrow = lane & 15;
    const bool hiHalf = (lane & 16) != 0;
    const int d0 = lane * 4;

    sh_w2[tid]  = w2v[tid];
    sh_w1l[tid] = w1last[tid];
    // padding rows 12..15 stay zero for the whole kernel
    float4 z4 = make_float4(0.f, 0.f, 0.f, 0.f);
#pragma unroll
    for (int s = S_NB; s < 16; ++s) *(float4*)&sh_neigh[wave][s][d0] = z4;
    __syncthreads();

#pragma unroll 1
    for (int g = 0; g < 4; ++g) {
        const int m = wave * 4 + g;
        const size_t rid = (size_t)blockIdx.x * 16 + m;
        const int b = (int)(rid / rows_per_b);

        // Close WAR window: previous iteration's DS reads of sh_neigh/sh_extra
        // must complete before the async engine rewrites those LDS rows.
        wait_ds0();

        // async global->LDS staging (each lane moves 16B per row)
        async_copy_b128(extra + (size_t)b * D_SZ + d0, &sh_extra[wave][d0]);
#pragma unroll
        for (int s = 0; s < S_NB; ++s) {
            size_t nr = neigh_idx ? (size_t)neigh_idx[rid * S_NB + s] : (rid * S_NB + s);
            async_copy_b128(neigh_base + nr * D_SZ + d0, &sh_neigh[wave][s][d0]);
        }
        if (lane < 16) sh_wrow[wave][lane] = (lane < S_NB) ? wgt[rid * S_NB + lane] : 0.f;
        wait_async0();

        // A fragments: A[M=s][K=k] = extra[k]*neigh[s][k], bf16
        v16bf af[4];
#pragma unroll
        for (int kc = 0; kc < 4; ++kc) {
            int blo = kc * 32 + (hiHalf ? 8 : 0);
#pragma unroll
            for (int e = 0; e < 8; ++e) {
                af[kc][e]     = (__bf16)(sh_neigh[wave][mrow][blo + e]      * sh_extra[wave][blo + e]);
                af[kc][e + 8] = (__bf16)(sh_neigh[wave][mrow][blo + 16 + e] * sh_extra[wave][blo + 16 + e]);
            }
        }
        float wr[8];
#pragma unroll
        for (int r = 0; r < 8; ++r) wr[r] = sh_wrow[wave][r + (hiHalf ? 8 : 0)];

        float part[8];
#pragma unroll
        for (int r = 0; r < 8; ++r) part[r] = 0.f;

        const int ncol = lane & 15;
#pragma unroll 1
        for (int t = 0; t < 8; ++t) {
            const int N = t * 16 + ncol;
            const float biasc = sh_w1l[N];
            v8f acc;
#pragma unroll
            for (int r = 0; r < 8; ++r) acc[r] = wr[r] * biasc;
#pragma unroll
            for (int kc = 0; kc < 4; ++kc) {
                v16bf bf = *(const v16bf*)(w1frag + ((size_t)(kc * 8 + t) * 32 + lane) * 16);
                acc = __builtin_amdgcn_wmma_f32_16x16x32_bf16(false, af[kc], false, bf,
                                                              (short)0, acc, false, false);
            }
            const float w2c = sh_w2[N];
#pragma unroll
            for (int r = 0; r < 8; ++r) {
                float v = acc[r];
                v = (v > 0.f) ? v : 0.2f * v;
                part[r] += v * w2c;
            }
        }
        // reduce across the 16-lane halves (masks < 16 stay in-half)
#pragma unroll
        for (int r = 0; r < 8; ++r) {
#pragma unroll
            for (int o = 1; o < 16; o <<= 1) part[r] += __shfl_xor(part[r], o, 32);
        }
        float oth[8];
#pragma unroll
        for (int r = 0; r < 8; ++r) oth[r] = __shfl_xor(part[r], 16, 32);

        float Lg[S_NB];
#pragma unroll
        for (int s = 0; s < S_NB; ++s) {
            int r = s & 7;
            Lg[s] = (s < 8) ? (hiHalf ? oth[r] : part[r]) : (hiHalf ? part[r] : oth[r]);
        }
        float mx = Lg[0];
#pragma unroll
        for (int s = 1; s < S_NB; ++s) mx = fmaxf(mx, Lg[s]);
        float al[S_NB], ssum = 0.f;
#pragma unroll
        for (int s = 0; s < S_NB; ++s) { al[s] = __expf(Lg[s] - mx); ssum += al[s]; }
        float inv = 1.f / ssum;

        // weighted neighbors -> concat[m][128..255]
#pragma unroll
        for (int j = 0; j < 4; ++j) {
            int d = d0 + j;
            float nv = 0.f;
#pragma unroll
            for (int s = 0; s < S_NB; ++s) nv += al[s] * sh_neigh[wave][s][d];
            sh_concat[m][D_SZ + d] = (__bf16)(nv * inv);
        }
        // self row -> concat[m][0..127]
        size_t srid = self_idx ? (size_t)self_idx[rid] : rid;
        float4 sv = *(const float4*)(self_base + srid * D_SZ + d0);
        sh_concat[m][d0 + 0] = (__bf16)sv.x;
        sh_concat[m][d0 + 1] = (__bf16)sv.y;
        sh_concat[m][d0 + 2] = (__bf16)sv.z;
        sh_concat[m][d0 + 3] = (__bf16)sv.w;
    }
    __syncthreads();   // publishes sh_concat (emits dscnt flush before barrier)

    // Phase 2: [16,256] @ w3 [256,128]; wave handles N-tiles {2*wave, 2*wave+1}
#pragma unroll 1
    for (int tt = 0; tt < 2; ++tt) {
        const int t2 = wave * 2 + tt;
        v8f acc;
#pragma unroll
        for (int r = 0; r < 8; ++r) acc[r] = 0.f;
#pragma unroll 1
        for (int kc = 0; kc < 8; ++kc) {
            int blo = kc * 32 + (hiHalf ? 8 : 0);
            v8bf lo = *(const v8bf*)&sh_concat[mrow][blo];
            v8bf hi = *(const v8bf*)&sh_concat[mrow][blo + 16];
            v16bf af2;
#pragma unroll
            for (int e = 0; e < 8; ++e) { af2[e] = lo[e]; af2[e + 8] = hi[e]; }
            v16bf bf = *(const v16bf*)(w3frag + ((size_t)(kc * 8 + t2) * 32 + lane) * 16);
            acc = __builtin_amdgcn_wmma_f32_16x16x32_bf16(false, af2, false, bf,
                                                          (short)0, acc, false, false);
        }
        const int col = t2 * 16 + (lane & 15);
#pragma unroll
        for (int r = 0; r < 8; ++r) {
            size_t row = (size_t)blockIdx.x * 16 + (hiHalf ? r + 8 : r);
            outp[row * D_SZ + col] = fmaxf(acc[r], 0.f);
        }
    }
}

// ===========================================================================
// final: out = h_local (already in out) + s_global / masksum[b]
// ===========================================================================
__global__ void finalize_kernel(float* __restrict__ out, const float* __restrict__ sg,
                                const float* __restrict__ msum, int n) {
    int t = blockIdx.x * blockDim.x + threadIdx.x;
    if (t < n) out[t] += sg[t] / msum[t / (L_SZ * D_SZ)];
}

// ===========================================================================
extern "C" void kernel_launch(void* const* d_in, const int* in_sizes, int n_in,
                              void* d_out, int out_size, void* d_ws, size_t ws_size,
                              hipStream_t stream) {
    const int*   inputs  = (const int*)  d_in[0];
    const int*   adj     = (const int*)  d_in[1];
    const int*   maskI   = (const int*)  d_in[2];
    const int*   item    = (const int*)  d_in[3];
    const int*   adj_all = (const int*)  d_in[4];
    const float* num_tab = (const float*)d_in[5];
    const float* emb     = (const float*)d_in[6];
    const float* a_loc   = (const float*)d_in[7];
    const float* gw1     = (const float*)d_in[8];
    const float* gw2     = (const float*)d_in[9];
    const float* gw3     = (const float*)d_in[10];
    float* out = (float*)d_out;

    char* ws = (char*)d_ws;
    size_t off = 0;
    auto take = [&](size_t bytes) -> char* {
        char* p = ws + off;
        off = (off + bytes + 255) & ~(size_t)255;
        return p;
    };
    const int N1 = B_SZ * L_SZ * S_NB;           // 61440
    const int N2 = N1 * S_NB;                    // 737280
    __bf16* w1frag = (__bf16*)take((size_t)HOP_N * 4 * 8 * 32 * 16 * sizeof(__bf16));
    __bf16* w3frag = (__bf16*)take((size_t)HOP_N * 8 * 8 * 32 * 16 * sizeof(__bf16));
    float*  sumit  = (float*) take((size_t)B_SZ * D_SZ * sizeof(float));
    float*  msum   = (float*) take((size_t)B_SZ * sizeof(float));
    int*    n1i    = (int*)   take((size_t)N1 * sizeof(int));
    float*  n1w    = (float*) take((size_t)N1 * sizeof(float));
    int*    n2i    = (int*)   take((size_t)N2 * sizeof(int));
    float*  n2w    = (float*) take((size_t)N2 * sizeof(float));
    float*  v0     = (float*) take((size_t)B_SZ * L_SZ * D_SZ * sizeof(float));
    float*  v1     = (float*) take((size_t)N1 * D_SZ * sizeof(float));
    float*  sg     = (float*) take((size_t)B_SZ * L_SZ * D_SZ * sizeof(float));

    // weight pre-swizzle to WMMA fragment layout (bf16)
    frag_pack_kernel<<<(HOP_N * 4 * 256 + 127) / 128, 128, 0, stream>>>(
        gw1, w1frag, 4, (D_SZ + 1) * D_SZ, HOP_N * 4 * 256);
    frag_pack_kernel<<<(HOP_N * 8 * 256 + 127) / 128, 128, 0, stream>>>(
        gw3, w3frag, 8, 2 * D_SZ * D_SZ, HOP_N * 8 * 256);

    sum_item_kernel<<<B_SZ, D_SZ, 0, stream>>>(emb, item, maskI, sumit, msum);

    expand_kernel<<<(N1 + 255) / 256, 256, 0, stream>>>(inputs, adj_all, num_tab, n1i, n1w, N1);
    expand_kernel<<<(N2 + 255) / 256, 256, 0, stream>>>(n1i, adj_all, num_tab, n2i, n2w, N2);

    local_agg_kernel<<<B_SZ, 256, 0, stream>>>(emb, inputs, adj, a_loc, out);

    const float* w1last0 = gw1 + (size_t)D_SZ * D_SZ;                       // hop0 bias row
    const float* w1last1 = gw1 + (size_t)(D_SZ + 1) * D_SZ + D_SZ * D_SZ;   // hop1 bias row
    const __bf16* w1f1 = w1frag + (size_t)4 * 8 * 32 * 16;
    const __bf16* w3f1 = w3frag + (size_t)8 * 8 * 32 * 16;

    // n_hop=0, hp=0: self=emb[inputs], neigh=emb[n1]
    global_agg_kernel<<<(B_SZ * L_SZ) / 16, 128, 0, stream>>>(
        emb, inputs, emb, n1i, n1w, sumit, w1frag, w1last0, gw2, w3frag, v0, L_SZ);
    // n_hop=0, hp=1: self=emb[n1], neigh=emb[n2]
    global_agg_kernel<<<N1 / 16, 128, 0, stream>>>(
        emb, n1i, emb, n2i, n2w, sumit, w1frag, w1last0, gw2, w3frag, v1, L_SZ * S_NB);
    // n_hop=1, hp=0: self=v0 (dense), neigh=v1 (dense)
    global_agg_kernel<<<(B_SZ * L_SZ) / 16, 128, 0, stream>>>(
        v0, nullptr, v1, nullptr, n1w, sumit, w1f1, w1last1, gw2 + D_SZ, w3f1, sg, L_SZ);

    finalize_kernel<<<(B_SZ * L_SZ * D_SZ + 255) / 256, 256, 0, stream>>>(
        out, sg, msum, B_SZ * L_SZ * D_SZ);
}
